// CAM_Module_2147483648305
// MI455X (gfx1250) — compile-verified, matrix operand order checked by reading, not run on hardware
//
#include <hip/hip_runtime.h>

typedef __attribute__((ext_vector_type(16))) _Float16 v16h;
typedef __attribute__((ext_vector_type(8)))  _Float16 v8h;
typedef __attribute__((ext_vector_type(4)))  _Float16 v4h;
typedef __attribute__((ext_vector_type(8)))  float    v8f;
typedef __attribute__((ext_vector_type(4)))  float    v4f;

#define LDSK 40   // 32 K-halves + 8 pad (80B row stride: 16B aligned, bank-conflict free)

// Load one 16x32 f16 WMMA fragment from LDS rows (row-major, stride LDSK halves).
// Element e of lane l holds (row = base + (l&15), K = 16*(e>>3) + 8*(l>>4) + (e&7)).
// Used for both A (rows = M) and B (rows = N, value = B[k][n]) fragments.
__device__ __forceinline__ v16h load_frag(const _Float16* rowp, int g) {
    union { v16h v; v8h h[2]; } u;
    u.h[0] = *reinterpret_cast<const v8h*>(rowp + 8 * g);        // K = 8g + 0..7
    u.h[1] = *reinterpret_cast<const v8h*>(rowp + 16 + 8 * g);   // K = 16 + 8g + 0..7
    return u.v;
}

// ---- K1 staging: 512 rows x 32 floats (one K-chunk of q) = 16 float4/thread ----
__device__ __forceinline__ void k1_issue_loads(const float* __restrict__ xb, int k0,
                                               int tid, v4f st[16]) {
    #pragma unroll
    for (int j = 0; j < 16; ++j) {
        int f   = tid + 256 * j;     // 0..4095
        int row = f >> 3;            // 8 float4 per row
        int c4  = f & 7;
        st[j] = ((const v4f*)(xb + (size_t)row * 4096 + k0))[c4];
    }
}
__device__ __forceinline__ void k1_store_lds(_Float16* buf, int tid, const v4f st[16]) {
    #pragma unroll
    for (int j = 0; j < 16; ++j) {
        int f   = tid + 256 * j;
        int row = f >> 3;
        int c4  = f & 7;
        v4h h;
        h[0] = (_Float16)st[j].x; h[1] = (_Float16)st[j].y;
        h[2] = (_Float16)st[j].z; h[3] = (_Float16)st[j].w;
        *(v4h*)&buf[row * LDSK + c4 * 4] = h;
    }
}

// ---------------------------------------------------------------------------
// Kernel 1: energy = q q^T (per batch, 512x512, K=4096) with fused softmax:
// softmax(rowmax - e) == exp(rowmin - e) / sum(exp(rowmin - e)).
// Block: 32 rows x 512 cols of energy. Grid: 16 batches * 16 strips = 256.
// Double-buffered LDS. Global loads issue above the WMMA block; an empty
// volatile-asm anchor on the staged registers forces their waits + f32->f16
// conversion BELOW the WMMA block, so load latency hides under compute.
// ---------------------------------------------------------------------------
__global__ __launch_bounds__(256) void cam_energy_softmax(
    const float* __restrict__ x, _Float16* __restrict__ att) {
    const int b     = blockIdx.x >> 4;
    const int strip = blockIdx.x & 15;
    const int m0    = strip * 32;

    const int tid  = threadIdx.x;
    const int wave = tid >> 5;
    const int lane = tid & 31;
    const int g    = lane >> 4;     // lane half (0: lanes 0-15, 1: lanes 16-31)
    const int ln   = lane & 15;
    const int rt   = wave & 1;      // row tile within strip (16 rows each)
    const int cg   = wave >> 1;     // column group (8 col-tiles = 128 cols each)

    __shared__ __align__(16) _Float16 lq[2][512 * LDSK]; // double-buffered q chunk (f16)
    __shared__ float red[4][32];                         // cross-wave row reductions

    const float* xb = x + (size_t)b * 512 * 4096;

    v8f acc[8];
    #pragma unroll
    for (int t = 0; t < 8; ++t)
        for (int i = 0; i < 8; ++i) acc[t][i] = 0.0f;

    v4f st[16];
    k1_issue_loads(xb, 0, tid, st);
    k1_store_lds(lq[0], tid, st);
    __syncthreads();

    for (int i = 0; i < 128; ++i) {              // 128 K-chunks of 32
        const int cur = i & 1;
        if (i + 1 < 128) k1_issue_loads(xb, (i + 1) * 32, tid, st);
        __builtin_amdgcn_sched_barrier(0);       // loads stay above compute

        const _Float16* lqc = lq[cur];
        v16h af = load_frag(&lqc[(m0 + rt * 16 + ln) * LDSK], g);
        v16h bf = load_frag(&lqc[(cg * 8 * 16 + ln) * LDSK], g);
        #pragma unroll
        for (int t = 0; t < 8; ++t) {
            v16h bn;
            if (t < 7) bn = load_frag(&lqc[(((cg * 8 + t + 1) * 16) + ln) * LDSK], g);
            acc[t] = __builtin_amdgcn_wmma_f32_16x16x32_f16(
                false, af, false, bf, (short)0, acc[t], false, false);
            if (t < 7) bf = bn;
        }

        // Anchor: consume staged loads here so s_wait_loadcnt + cvt land
        // after the WMMAs (pure fptrunc cannot float back above this).
        if (i + 1 < 128) {
            #pragma unroll
            for (int j = 0; j < 16; ++j) asm volatile("" : "+v"(st[j]));
            k1_store_lds(lq[cur ^ 1], tid, st);
        }
        __syncthreads();
    }

    // -------- fused softmax over each 512-wide energy row --------
    const int rrow = rt * 16 + g * 8;   // first of the 8 rows this lane-half holds

    // row minimum
    float rmin[8];
    #pragma unroll
    for (int v = 0; v < 8; ++v) {
        float m = acc[0][v];
        #pragma unroll
        for (int t = 1; t < 8; ++t) m = fminf(m, acc[t][v]);
        for (int off = 1; off < 16; off <<= 1)
            m = fminf(m, __shfl_xor(m, off, 32));
        rmin[v] = m;
    }
    if (ln == 0)
        for (int v = 0; v < 8; ++v) red[cg][rrow + v] = rmin[v];
    __syncthreads();
    #pragma unroll
    for (int v = 0; v < 8; ++v)
        rmin[v] = fminf(fminf(red[0][rrow + v], red[1][rrow + v]),
                        fminf(red[2][rrow + v], red[3][rrow + v]));
    __syncthreads();

    // p = exp(min - e), row sum
    float rsum[8];
    #pragma unroll
    for (int v = 0; v < 8; ++v) rsum[v] = 0.0f;
    #pragma unroll
    for (int t = 0; t < 8; ++t)
        #pragma unroll
        for (int v = 0; v < 8; ++v) {
            float e = __expf(rmin[v] - acc[t][v]);
            acc[t][v] = e;
            rsum[v] += e;
        }
    #pragma unroll
    for (int v = 0; v < 8; ++v)
        for (int off = 1; off < 16; off <<= 1)
            rsum[v] += __shfl_xor(rsum[v], off, 32);
    if (ln == 0)
        for (int v = 0; v < 8; ++v) red[cg][rrow + v] = rsum[v];
    __syncthreads();
    float inv[8];
    #pragma unroll
    for (int v = 0; v < 8; ++v)
        inv[v] = 1.0f / (red[0][rrow + v] + red[1][rrow + v] +
                         red[2][rrow + v] + red[3][rrow + v]);

    // store attention as f16 (row-major 512x512 per batch)
    _Float16* attb = att + (size_t)b * 512 * 512;
    #pragma unroll
    for (int t = 0; t < 8; ++t) {
        int col = (cg * 8 + t) * 16 + ln;
        #pragma unroll
        for (int v = 0; v < 8; ++v) {
            int row = m0 + rrow + v;
            attb[(size_t)row * 512 + col] = (_Float16)(acc[t][v] * inv[v]);
        }
    }
}

// ---------------------------------------------------------------------------
// Kernel 2: out = gamma * (att @ q) + x.  M=512, N=4096, K=512 per batch.
// Block: 64 x 256 output tile. Grid: (16 n-strips, 8 m-strips, 16 batches).
// Same double-buffered + anchored-staging scheme.
// ---------------------------------------------------------------------------
__global__ __launch_bounds__(256) void cam_apply(
    const float* __restrict__ x, const _Float16* __restrict__ att,
    const float* __restrict__ gamma, float* __restrict__ out) {
    const int b  = blockIdx.z;
    const int m0 = blockIdx.y * 64;
    const int n0 = blockIdx.x * 256;

    const int tid  = threadIdx.x;
    const int wave = tid >> 5;
    const int lane = tid & 31;
    const int g    = lane >> 4;
    const int ln   = lane & 15;
    const int rt   = wave & 3;     // 4 row tiles (16 rows each)
    const int cg   = wave >> 2;    // 2 col groups (8 col-tiles = 128 cols each)

    __shared__ __align__(16) _Float16 lA[2][64 * LDSK];   // att[m0:m0+64, k0:k0+32]
    __shared__ __align__(16) _Float16 lB[2][256 * LDSK];  // lB[n][k] = q[k0+k, n0+n]

    const float*        xb   = x + (size_t)b * 512 * 4096;
    const _Float16*     attb = att + (size_t)b * 512 * 512;
    const unsigned int* As   = (const unsigned int*)attb;

    v8f acc[8];
    #pragma unroll
    for (int t = 0; t < 8; ++t)
        for (int i = 0; i < 8; ++i) acc[t][i] = 0.0f;

    unsigned int stA[4];
    v4f          stB[8];

    auto issueAB = [&](int k0) {
        #pragma unroll
        for (int j = 0; j < 4; ++j) {
            int f = tid + 256 * j;          // 1024 dwords: 64 rows x 16 dwords
            int row = f >> 4;
            int d   = f & 15;
            stA[j] = As[(size_t)(m0 + row) * 256 + (k0 >> 1) + d];
        }
        #pragma unroll
        for (int j = 0; j < 8; ++j) {
            int f  = tid + 256 * j;         // 2048 float4: 32 k-rows x 64 float4
            int k  = f >> 6;
            int c4 = f & 63;
            stB[j] = ((const v4f*)(xb + (size_t)(k0 + k) * 4096 + n0))[c4];
        }
    };
    auto storeAB = [&](int buf) {
        #pragma unroll
        for (int j = 0; j < 4; ++j) {
            int f = tid + 256 * j;
            int row = f >> 4;
            int d   = f & 15;
            *(unsigned int*)&lA[buf][row * LDSK + d * 2] = stA[j];
        }
        #pragma unroll
        for (int j = 0; j < 8; ++j) {
            int f  = tid + 256 * j;
            int k  = f >> 6;
            int c4 = f & 63;
            int n  = c4 * 4;
            lB[buf][(n + 0) * LDSK + k] = (_Float16)stB[j].x;
            lB[buf][(n + 1) * LDSK + k] = (_Float16)stB[j].y;
            lB[buf][(n + 2) * LDSK + k] = (_Float16)stB[j].z;
            lB[buf][(n + 3) * LDSK + k] = (_Float16)stB[j].w;
        }
    };

    issueAB(0);
    storeAB(0);
    __syncthreads();

    for (int i = 0; i < 16; ++i) {               // 16 K-chunks of 32
        const int cur = i & 1;
        if (i + 1 < 16) issueAB((i + 1) * 32);
        __builtin_amdgcn_sched_barrier(0);       // loads stay above compute

        v16h af = load_frag(&lA[cur][(rt * 16 + ln) * LDSK], g);
        v16h bf = load_frag(&lB[cur][(cg * 8 * 16 + ln) * LDSK], g);
        #pragma unroll
        for (int t = 0; t < 8; ++t) {
            v16h bn;
            if (t < 7) bn = load_frag(&lB[cur][(((cg * 8 + t + 1) * 16) + ln) * LDSK], g);
            acc[t] = __builtin_amdgcn_wmma_f32_16x16x32_f16(
                false, af, false, bf, (short)0, acc[t], false, false);
            if (t < 7) bf = bn;
        }

        if (i + 1 < 16) {
            #pragma unroll
            for (int j = 0; j < 4; ++j) asm volatile("" : "+v"(stA[j]));
            #pragma unroll
            for (int j = 0; j < 8; ++j) asm volatile("" : "+v"(stB[j]));
            storeAB(cur ^ 1);
        }
        __syncthreads();
    }

    // epilogue: out = gamma * acc + x
    const float gm = gamma[0];
    #pragma unroll
    for (int t = 0; t < 8; ++t) {
        int N = n0 + (cg * 8 + t) * 16 + ln;
        #pragma unroll
        for (int v = 0; v < 8; ++v) {
            int M = m0 + rt * 16 + g * 8 + v;
            size_t idx = ((size_t)b * 512 + M) * 4096 + N;
            out[idx] = gm * acc[t][v] + x[idx];
        }
    }
}

extern "C" void kernel_launch(void* const* d_in, const int* in_sizes, int n_in,
                              void* d_out, int out_size, void* d_ws, size_t ws_size,
                              hipStream_t stream) {
    const float* x     = (const float*)d_in[0];
    const float* gamma = (const float*)d_in[1];
    float*       out   = (float*)d_out;
    _Float16*    att   = (_Float16*)d_ws;   // 16*512*512 f16 = 8.4 MB

    // Energy + fused softmax -> attention (f16) in workspace
    cam_energy_softmax<<<dim3(256), dim3(256), 0, stream>>>(x, att);
    // out = gamma * (att @ q) + x
    cam_apply<<<dim3(16, 8, 16), dim3(256), 0, stream>>>(x, att, gamma, out);
}
